// TemporalTransformer_83047487635862
// MI455X (gfx1250) — matrix operand order repaired and checked by reading, not compiled, not run on hardware
//
#include <hip/hip_runtime.h>
#include <hip/hip_bf16.h>
#include <stdint.h>

typedef __attribute__((ext_vector_type(16))) _Float16 v16h;
typedef __attribute__((ext_vector_type(8)))  float    v8f;
typedef int v4i_ __attribute__((vector_size(4 * sizeof(int))));
typedef _Float16 half_t;

#define B_    4
#define S_    1024
#define IN_D_ 16
#define OUT_D_ 16
#define QD_   32
#define HID_  256
#define HEADS_ 8
#define HD_   32
#define L_    4
#define TED_  256
#define DECAY_ 0.1f
#define SCALE_ 0.17677669529663687f  /* 32^-0.5 */

// CDNA5 async global->LDS copy path (ASYNCcnt-tracked), with sync fallback.
#if defined(__has_builtin)
#  if __has_builtin(__builtin_amdgcn_global_load_async_to_lds_b128) && \
      __has_builtin(__builtin_amdgcn_s_wait_asynccnt)
#    define USE_ASYNC_LDS 1
#  endif
#endif
#ifndef USE_ASYNC_LDS
#  define USE_ASYNC_LDS 0
#endif

#if USE_ASYNC_LDS
#define ASYNC_CP16(gp, lp)                                                    \
    __builtin_amdgcn_global_load_async_to_lds_b128((v4i_*)(gp), (v4i_*)(lp), 0, 0)
#define ASYNC_WAIT() __builtin_amdgcn_s_wait_asynccnt(0)
#else
#define ASYNC_WAIT()
#endif

union Frag16 { v16h h; uint32_t u[8]; };

__device__ __forceinline__ v8f wmma_f16(v16h a, v16h b, v8f c) {
    return __builtin_amdgcn_wmma_f32_16x16x32_f16(false, a, false, b, (short)0, c, false, false);
}

__device__ __forceinline__ float gelu_f(float x) {
    return 0.5f * x * (1.0f + erff(x * 0.70710678118654752f));
}

// ---------------------------------------------------------------------------
// f32 -> f16 weight conversion
// ---------------------------------------------------------------------------
__global__ void cvt_f16_kernel(const float* __restrict__ src, half_t* __restrict__ dst, int n) {
    int i = blockIdx.x * blockDim.x + threadIdx.x;
    if (i < n) dst[i] = (half_t)src[i];
}

// ---------------------------------------------------------------------------
// fused decay * rel_bias table:  db[l, rel+S-1, h] = exp(-DECAY*|rel|)*bias
// ---------------------------------------------------------------------------
__global__ void decbias_kernel(const float* __restrict__ rel_bias, float* __restrict__ db, int n) {
    int i = blockIdx.x * blockDim.x + threadIdx.x;
    if (i >= n) return;
    int rel = (i / HEADS_) % (2 * S_ - 1) - (S_ - 1);
    db[i] = __expf(-DECAY_ * fabsf((float)rel)) * rel_bias[i];
}

// ---------------------------------------------------------------------------
// time MLP + query MLP (one block per batch, 256 threads)
// ---------------------------------------------------------------------------
__global__ __launch_bounds__(256) void time_query_kernel(
    const float* __restrict__ time, const float* __restrict__ tm_w1, const float* __restrict__ tm_b1,
    const float* __restrict__ tm_w2, const float* __restrict__ tm_b2,
    const float* __restrict__ query, const float* __restrict__ q_w1, const float* __restrict__ q_b1,
    const float* __restrict__ q_w2, const float* __restrict__ q_b2,
    float* __restrict__ te, float* __restrict__ qe)
{
    int b = blockIdx.x;
    int t = threadIdx.x;
    __shared__ float emb[TED_];
    __shared__ float hid[TED_ * 4];

    {   // sinusoidal embedding: half=128, freq_j = exp(-ln(10000)*j/127)
        int j = t & 127;
        float fr = __expf(-9.210340371976184f * (float)j / 127.0f);
        float a = time[b] * fr;
        emb[t] = (t < 128) ? __sinf(a) : __cosf(a);
    }
    __syncthreads();
    for (int k = t; k < TED_ * 4; k += 256) {
        float s = tm_b1[k];
        for (int j = 0; j < TED_; ++j) s += tm_w1[k * TED_ + j] * emb[j];
        float sp = (s > 20.0f) ? s : log1pf(__expf(s));   // softplus
        hid[k] = s * tanhf(sp);                           // mish
    }
    __syncthreads();
    {
        float s = tm_b2[t];
        for (int k = 0; k < TED_ * 4; ++k) s += tm_w2[t * TED_ * 4 + k] * hid[k];
        te[b * TED_ + t] = s;
    }
    __syncthreads();
    {
        float s = q_b1[t];
        for (int j = 0; j < QD_; ++j) s += q_w1[t * QD_ + j] * query[b * QD_ + j];
        hid[t] = gelu_f(s);
    }
    __syncthreads();
    {
        float s = q_b2[t];
        for (int k = 0; k < HID_; ++k) s += q_w2[t * HID_ + k] * hid[k];
        qe[b * HID_ + t] = s;
    }
}

// ---------------------------------------------------------------------------
// per-layer time conditioning: tp = Lin(gelu(Lin(te)))
// ---------------------------------------------------------------------------
__global__ __launch_bounds__(256) void time_cond_kernel(
    const float* __restrict__ te, const float* __restrict__ t1w, const float* __restrict__ t1b,
    const float* __restrict__ t2w, const float* __restrict__ t2b, float* __restrict__ tp)
{
    int b = blockIdx.x;
    int t = threadIdx.x;
    __shared__ float hid[HID_];
    float s = t1b[t];
    for (int j = 0; j < TED_; ++j) s += t1w[t * TED_ + j] * te[b * TED_ + j];
    hid[t] = gelu_f(s);
    __syncthreads();
    float s2 = t2b[t];
    for (int k = 0; k < HID_; ++k) s2 += t2w[t * HID_ + k] * hid[k];
    tp[b * HID_ + t] = s2;
}

// ---------------------------------------------------------------------------
// input projection + query conditioning: h = x @ in_w^T + in_b + qe
// ---------------------------------------------------------------------------
__global__ __launch_bounds__(256) void input_proj_kernel(
    const float* __restrict__ x, const float* __restrict__ in_w, const float* __restrict__ in_b,
    const float* __restrict__ qe, float* __restrict__ h)
{
    int row = blockIdx.x;          // b*S + s
    int t = threadIdx.x;
    int b = row >> 10;
    const float* xr = x + (size_t)row * IN_D_;
    float s = in_b[t];
    #pragma unroll
    for (int j = 0; j < IN_D_; ++j) s += in_w[t * IN_D_ + j] * xr[j];
    h[(size_t)row * HID_ + t] = s + qe[b * HID_ + t];
}

// ---------------------------------------------------------------------------
// LayerNorm(h [+ tp]) * gamma + beta -> f16
// ---------------------------------------------------------------------------
__global__ __launch_bounds__(256) void ln_kernel(
    const float* __restrict__ h, const float* __restrict__ tp,
    const float* __restrict__ gw, const float* __restrict__ gb,
    half_t* __restrict__ out)
{
    int row = blockIdx.x;
    int b = row >> 10;
    int t = threadIdx.x;
    __shared__ float red[256];
    float x = h[(size_t)row * HID_ + t] + (tp ? tp[b * HID_ + t] : 0.0f);
    red[t] = x;
    __syncthreads();
    for (int s = 128; s > 0; s >>= 1) { if (t < s) red[t] += red[t + s]; __syncthreads(); }
    float mean = red[0] * (1.0f / HID_);
    __syncthreads();
    float d = x - mean;
    red[t] = d * d;
    __syncthreads();
    for (int s = 128; s > 0; s >>= 1) { if (t < s) red[t] += red[t + s]; __syncthreads(); }
    float var = red[0] * (1.0f / HID_);
    float xn = d * rsqrtf(var + 1e-5f);
    out[(size_t)row * HID_ + t] = (half_t)(xn * gw[t] + gb[t]);
}

// ---------------------------------------------------------------------------
// generic WMMA GEMM: C[M,N] = A[M,K](f16) @ W[N,K]^T(f16) + bias
// Tile 128x64, BK=32, 256 threads = 8 waves, each wave computes 32x32.
// Double-buffered LDS; tiles staged with async global->LDS copies on CDNA5.
// EPI: 1 = gelu -> f16 out ; 2 = accumulate into f32 buffer ; 3 = QKV scatter
// ---------------------------------------------------------------------------
template <int EPI>
__global__ __launch_bounds__(256) void gemm_wmma_kernel(
    const half_t* __restrict__ A, const half_t* __restrict__ W,
    const float* __restrict__ bias,
    float* __restrict__ Cf32, half_t* __restrict__ Ch,
    half_t* __restrict__ Qo, half_t* __restrict__ Ko, half_t* __restrict__ Vto,
    int M, int N, int Kd)
{
    __shared__ half_t sA[2][128][32];
    __shared__ half_t sB[2][64][32];

    int tile_n = blockIdx.x * 64;
    int tile_m = blockIdx.y * 128;
    int t = threadIdx.x;
    int wave = t >> 5, lane = t & 31;
    int wm = (wave >> 1) * 32, wn = (wave & 1) * 32;
    int mL = lane & 15, hiL = lane >> 4;

    int arow = t >> 1, acol = (t & 1) * 16;   // A: 128 rows x 32 halfs (2x b128/thread)
    int brow = t >> 2, bcol = (t & 3) * 8;    // B: 64 rows x 32 halfs  (1x b128/thread)

    const half_t* Abase = A + (size_t)(tile_m + arow) * Kd + acol;
    const half_t* Bbase = W + (size_t)(tile_n + brow) * Kd + bcol;

    auto stage = [&](int k0, int buf) {
        const half_t* ap = Abase + k0;
        const half_t* bp = Bbase + k0;
#if USE_ASYNC_LDS
        ASYNC_CP16(ap,     &sA[buf][arow][acol]);
        ASYNC_CP16(ap + 8, &sA[buf][arow][acol + 8]);
        ASYNC_CP16(bp,     &sB[buf][brow][bcol]);
#else
        *(float4*)&sA[buf][arow][acol]     = *(const float4*)ap;
        *(float4*)&sA[buf][arow][acol + 8] = *(const float4*)(ap + 8);
        *(float4*)&sB[buf][brow][bcol]     = *(const float4*)bp;
#endif
    };

    v8f acc[2][2] = {{{}, {}}, {{}, {}}};

    int nstages = Kd >> 5;
    stage(0, 0);
    ASYNC_WAIT();
    __syncthreads();

    int buf = 0;
    for (int s = 0; s < nstages; ++s) {
        if (s + 1 < nstages) stage((s + 1) << 5, buf ^ 1);

        Frag16 a0, a1, b0, b1;
        #pragma unroll
        for (int j = 0; j < 8; ++j) {
            int ka = (j < 4 ? 2 * j : 8 + 2 * j) + 8 * hiL;  // A-layout K pair base
            a0.u[j] = *(const uint32_t*)&sA[buf][wm + mL][ka];
            a1.u[j] = *(const uint32_t*)&sA[buf][wm + 16 + mL][ka];
            int kb = (j < 4 ? 2 * j : 8 + 2 * j) + 8 * hiL;  // same K pair base as A
            b0.u[j] = *(const uint32_t*)&sB[buf][wn + mL][kb];
            b1.u[j] = *(const uint32_t*)&sB[buf][wn + 16 + mL][kb];
        }
        acc[0][0] = wmma_f16(a0.h, b0.h, acc[0][0]);
        acc[0][1] = wmma_f16(a0.h, b1.h, acc[0][1]);
        acc[1][0] = wmma_f16(a1.h, b0.h, acc[1][0]);
        acc[1][1] = wmma_f16(a1.h, b1.h, acc[1][1]);

        if (s + 1 < nstages) {
            ASYNC_WAIT();
            __syncthreads();
            buf ^= 1;
        }
    }

    #pragma unroll
    for (int mi = 0; mi < 2; ++mi)
    #pragma unroll
    for (int ni = 0; ni < 2; ++ni) {
        v8f a = acc[mi][ni];
        #pragma unroll
        for (int r = 0; r < 8; ++r) {
            int gm = tile_m + wm + mi * 16 + r + 8 * hiL;
            int gn = tile_n + wn + ni * 16 + mL;
            float val = a[r] + bias[gn];
            if (EPI == 1) {
                Ch[(size_t)gm * N + gn] = (half_t)gelu_f(val);
            } else if (EPI == 2) {
                Cf32[(size_t)gm * N + gn] += val;
            } else if (EPI == 3) {
                int b = gm >> 10, s = gm & 1023;
                int which = gn >> 8, nn = gn & 255;
                int hh = nn >> 5, dd = nn & 31;
                half_t hv = (half_t)val;
                size_t bh = (size_t)(b * HEADS_ + hh);
                if (which == 0)      Qo[(bh * S_ + s) * HD_ + dd] = hv;
                else if (which == 1) Ko[(bh * S_ + s) * HD_ + dd] = hv;
                else                 Vto[(bh * HD_ + dd) * S_ + s] = hv;  // V^T [b,h,hd,s]
            }
        }
    }
}

// ---------------------------------------------------------------------------
// flash attention: block = 4 waves, each wave owns one 16-row query tile of
// one (b, head). Streams 32-key chunks: 2 score WMMAs + online softmax +
// C-layout -> A-layout P transpose through LDS + 2 output WMMAs.
// Bias uses the precomputed fused exp(-decay|rel|)*rel_bias table.
// ---------------------------------------------------------------------------
__global__ __launch_bounds__(128) void attn_kernel(
    const half_t* __restrict__ Q, const half_t* __restrict__ K,
    const half_t* __restrict__ Vt, const float* __restrict__ db,
    half_t* __restrict__ O)
{
    int bh = blockIdx.x;                 // 0..31
    int b = bh >> 3, hh = bh & 7;
    int wave = threadIdx.x >> 5, lane = threadIdx.x & 31;
    int q0 = (blockIdx.y * 4 + wave) * 16;
    int mL = lane & 15, hiL = lane >> 4;

    const half_t* qp = Q  + (size_t)bh * S_ * HD_;
    const half_t* kp = K  + (size_t)bh * S_ * HD_;
    const half_t* vp = Vt + (size_t)bh * HD_ * S_;
    const float*  dbh = db + hh;         // index by (rel + S-1)*HEADS_

    __shared__ half_t sP[4][16][32];

    Frag16 qa;
    #pragma unroll
    for (int j = 0; j < 8; ++j) {
        int ka = (j < 4 ? 2 * j : 8 + 2 * j) + 8 * hiL;
        qa.u[j] = *(const uint32_t*)&qp[(q0 + mL) * HD_ + ka];
    }

    v8f o0 = {}, o1 = {};
    float rmax[8], rsum[8];
    #pragma unroll
    for (int r = 0; r < 8; ++r) { rmax[r] = -1e30f; rsum[r] = 0.0f; }

    for (int kc = 0; kc < S_; kc += 32) {
        // K^T fragments: B[kdim][key], lane key = mL (+16), kdim = A-layout pair base
        Frag16 kb0, kb1;
        #pragma unroll
        for (int j = 0; j < 8; ++j) {
            int kd = (j < 4 ? 2 * j : 8 + 2 * j) + 8 * hiL;
            kb0.u[j] = *(const uint32_t*)&kp[(kc + mL) * HD_ + kd];
            kb1.u[j] = *(const uint32_t*)&kp[(kc + 16 + mL) * HD_ + kd];
        }
        v8f c0 = {}, c1 = {};
        c0 = wmma_f16(qa.h, kb0.h, c0);
        c1 = wmma_f16(qa.h, kb1.h, c1);

        float p0[8], p1[8], fac[8];
        #pragma unroll
        for (int r = 0; r < 8; ++r) {
            int qi = q0 + r + 8 * hiL;
            int rel0 = qi - (kc + mL), rel1 = rel0 - 16;
            float s0 = c0[r] * SCALE_ + dbh[(rel0 + S_ - 1) * HEADS_];
            float s1 = c1[r] * SCALE_ + dbh[(rel1 + S_ - 1) * HEADS_];
            float cm = fmaxf(s0, s1);
            #pragma unroll
            for (int d = 1; d < 16; d <<= 1) cm = fmaxf(cm, __shfl_xor(cm, d, 32));
            float nm = fmaxf(rmax[r], cm);
            float f = __expf(rmax[r] - nm);
            rmax[r] = nm;
            float e0 = __expf(s0 - nm), e1 = __expf(s1 - nm);
            float cs = e0 + e1;
            #pragma unroll
            for (int d = 1; d < 16; d <<= 1) cs += __shfl_xor(cs, d, 32);
            rsum[r] = rsum[r] * f + cs;
            fac[r] = f; p0[r] = e0; p1[r] = e1;
        }
        #pragma unroll
        for (int r = 0; r < 8; ++r) { o0[r] *= fac[r]; o1[r] *= fac[r]; }

        // P: C-layout -> LDS -> A-layout
        #pragma unroll
        for (int r = 0; r < 8; ++r) {
            int m = r + 8 * hiL;
            sP[wave][m][mL]      = (half_t)p0[r];
            sP[wave][m][16 + mL] = (half_t)p1[r];
        }
        __syncthreads();
        Frag16 pa;
        #pragma unroll
        for (int j = 0; j < 8; ++j) {
            int ka = (j < 4 ? 2 * j : 8 + 2 * j) + 8 * hiL;
            pa.u[j] = *(const uint32_t*)&sP[wave][mL][ka];
        }
        __syncthreads();

        // V fragments from transposed V: B[key][hdcol], lane col = mL (+16)
        Frag16 vb0, vb1;
        #pragma unroll
        for (int j = 0; j < 8; ++j) {
            int kk = kc + (j < 4 ? 2 * j : 8 + 2 * j) + 8 * hiL;
            vb0.u[j] = *(const uint32_t*)&vp[(size_t)mL * S_ + kk];
            vb1.u[j] = *(const uint32_t*)&vp[(size_t)(16 + mL) * S_ + kk];
        }
        o0 = wmma_f16(pa.h, vb0.h, o0);
        o1 = wmma_f16(pa.h, vb1.h, o1);
    }

    #pragma unroll
    for (int r = 0; r < 8; ++r) {
        int qi = q0 + r + 8 * hiL;
        float inv = 1.0f / rsum[r];
        size_t base = ((size_t)(b * S_ + qi)) * HID_ + hh * HD_;
        O[base + mL]      = (half_t)(o0[r] * inv);
        O[base + 16 + mL] = (half_t)(o1[r] * inv);
    }
}

// ---------------------------------------------------------------------------
// final projection: out = h @ out_w^T + out_b   (HID -> 16)
// ---------------------------------------------------------------------------
__global__ __launch_bounds__(256) void out_proj_kernel(
    const float* __restrict__ h, const float* __restrict__ out_w,
    const float* __restrict__ out_b, float* __restrict__ out)
{
    int idx = blockIdx.x * blockDim.x + threadIdx.x;    // B*S*OUT_D
    if (idx >= B_ * S_ * OUT_D_) return;
    int o = idx & 15, row = idx >> 4;
    float s = out_b[o];
    for (int k = 0; k < HID_; ++k) s += out_w[o * HID_ + k] * h[(size_t)row * HID_ + k];
    out[idx] = s;
}

// ---------------------------------------------------------------------------
extern "C" void kernel_launch(void* const* d_in, const int* in_sizes, int n_in,
                              void* d_out, int out_size, void* d_ws, size_t ws_size,
                              hipStream_t stream) {
    const float* x      = (const float*)d_in[0];
    const float* query  = (const float*)d_in[1];
    const float* timev  = (const float*)d_in[2];
    const float* tm_w1  = (const float*)d_in[3];
    const float* tm_b1  = (const float*)d_in[4];
    const float* tm_w2  = (const float*)d_in[5];
    const float* tm_b2  = (const float*)d_in[6];
    const float* in_w   = (const float*)d_in[7];
    const float* in_b   = (const float*)d_in[8];
    const float* q_w1   = (const float*)d_in[9];
    const float* q_b1   = (const float*)d_in[10];
    const float* q_w2   = (const float*)d_in[11];
    const float* q_b2   = (const float*)d_in[12];
    const float* n1_w   = (const float*)d_in[13];
    const float* n1_b   = (const float*)d_in[14];
    const float* qkv_w  = (const float*)d_in[15];
    const float* qkv_b  = (const float*)d_in[16];
    const float* ap_w   = (const float*)d_in[17];
    const float* ap_b   = (const float*)d_in[18];
    const float* rel_b  = (const float*)d_in[19];
    const float* n2_w   = (const float*)d_in[20];
    const float* n2_b   = (const float*)d_in[21];
    const float* f1_w   = (const float*)d_in[22];
    const float* f1_b   = (const float*)d_in[23];
    const float* f2_w   = (const float*)d_in[24];
    const float* f2_b   = (const float*)d_in[25];
    const float* t1_w   = (const float*)d_in[26];
    const float* t1_b   = (const float*)d_in[27];
    const float* t2_w   = (const float*)d_in[28];
    const float* t2_b   = (const float*)d_in[29];
    const float* out_w  = (const float*)d_in[30];
    const float* out_b  = (const float*)d_in[31];

    const int MS = B_ * S_;   // 4096 rows
    const int NREL = L_ * (2 * S_ - 1) * HEADS_;

    // --- workspace layout (256B aligned) ---
    uint8_t* ws = (uint8_t*)d_ws;
    size_t off = 0;
    auto take = [&](size_t bytes) -> uint8_t* {
        uint8_t* p = ws + off;
        off = (off + bytes + 255) & ~(size_t)255;
        return p;
    };
    float*  h_f   = (float*) take((size_t)MS * HID_ * 4);
    float*  te    = (float*) take(B_ * TED_ * 4);
    float*  qe    = (float*) take(B_ * HID_ * 4);
    float*  tp    = (float*) take(B_ * HID_ * 4);
    float*  db    = (float*) take((size_t)NREL * 4);
    half_t* xn    = (half_t*)take((size_t)MS * HID_ * 2);
    half_t* qh    = (half_t*)take((size_t)MS * HID_ * 2);
    half_t* kh    = (half_t*)take((size_t)MS * HID_ * 2);
    half_t* vth   = (half_t*)take((size_t)MS * HID_ * 2);
    half_t* oh    = (half_t*)take((size_t)MS * HID_ * 2);
    half_t* gh    = (half_t*)take((size_t)MS * 4 * HID_ * 2);
    half_t* qkv16 = (half_t*)take((size_t)L_ * 3 * HID_ * HID_ * 2);
    half_t* ap16  = (half_t*)take((size_t)L_ * HID_ * HID_ * 2);
    half_t* f116  = (half_t*)take((size_t)L_ * 4 * HID_ * HID_ * 2);
    half_t* f216  = (half_t*)take((size_t)L_ * HID_ * 4 * HID_ * 2);

    auto cvt = [&](const float* src, half_t* dst, int n) {
        cvt_f16_kernel<<<(n + 255) / 256, 256, 0, stream>>>(src, dst, n);
    };
    cvt(qkv_w, qkv16, L_ * 3 * HID_ * HID_);
    cvt(ap_w,  ap16,  L_ * HID_ * HID_);
    cvt(f1_w,  f116,  L_ * 4 * HID_ * HID_);
    cvt(f2_w,  f216,  L_ * HID_ * 4 * HID_);
    decbias_kernel<<<(NREL + 255) / 256, 256, 0, stream>>>(rel_b, db, NREL);

    time_query_kernel<<<B_, 256, 0, stream>>>(timev, tm_w1, tm_b1, tm_w2, tm_b2,
                                              query, q_w1, q_b1, q_w2, q_b2, te, qe);
    input_proj_kernel<<<MS, 256, 0, stream>>>(x, in_w, in_b, qe, h_f);

    for (int l = 0; l < L_; ++l) {
        time_cond_kernel<<<B_, 256, 0, stream>>>(
            te, t1_w + (size_t)l * HID_ * TED_, t1_b + l * HID_,
                t2_w + (size_t)l * HID_ * HID_, t2_b + l * HID_, tp);

        ln_kernel<<<MS, 256, 0, stream>>>(h_f, tp, n1_w + l * HID_, n1_b + l * HID_, xn);

        // QKV: [4096,256] x [768,256]^T
        gemm_wmma_kernel<3><<<dim3(3 * HID_ / 64, MS / 128), 256, 0, stream>>>(
            xn, qkv16 + (size_t)l * 3 * HID_ * HID_, qkv_b + l * 3 * HID_,
            nullptr, nullptr, qh, kh, vth, MS, 3 * HID_, HID_);

        attn_kernel<<<dim3(B_ * HEADS_, S_ / 16 / 4), 128, 0, stream>>>(
            qh, kh, vth, db + (size_t)l * (2 * S_ - 1) * HEADS_, oh);

        // h += o @ ap^T + ap_b
        gemm_wmma_kernel<2><<<dim3(HID_ / 64, MS / 128), 256, 0, stream>>>(
            oh, ap16 + (size_t)l * HID_ * HID_, ap_b + l * HID_,
            h_f, nullptr, nullptr, nullptr, nullptr, MS, HID_, HID_);

        ln_kernel<<<MS, 256, 0, stream>>>(h_f, nullptr, n2_w + l * HID_, n2_b + l * HID_, xn);

        // g = gelu(xn @ f1^T + b) : [4096,256] x [1024,256]^T
        gemm_wmma_kernel<1><<<dim3(4 * HID_ / 64, MS / 128), 256, 0, stream>>>(
            xn, f116 + (size_t)l * 4 * HID_ * HID_, f1_b + l * 4 * HID_,
            nullptr, gh, nullptr, nullptr, nullptr, MS, 4 * HID_, HID_);

        // h += g @ f2^T + b : [4096,1024] x [256,1024]^T
        gemm_wmma_kernel<2><<<dim3(HID_ / 64, MS / 128), 256, 0, stream>>>(
            gh, f216 + (size_t)l * HID_ * 4 * HID_, f2_b + l * HID_,
            h_f, nullptr, nullptr, nullptr, nullptr, MS, HID_, 4 * HID_);
    }

    out_proj_kernel<<<(MS * OUT_D_ + 255) / 256, 256, 0, stream>>>(
        h_f, out_w, out_b, (float*)d_out);
}
